// LinearQwen3MoeSparseMoeBlock_5025111737026
// MI455X (gfx1250) — matrix operand order repaired and checked by reading, not compiled, not run on hardware
//
#include <hip/hip_runtime.h>
#include <hip/hip_bf16.h>

// ---- problem constants (from reference) ----
#define BQ 2
#define SQ 1024
#define TT (BQ * SQ)   // 2048 tokens
#define HH 2048        // hidden
#define EE 8           // experts
#define II 768         // intermediate

#define LDS_STRIDE 40  // halves per LDS tile row (20 words; bank-conflict-free,
                       // 16B-aligned row base for b128 fragment loads)

typedef unsigned short u16;
typedef unsigned int   u32;

typedef __attribute__((ext_vector_type(16))) __bf16 v16bf;
typedef __attribute__((ext_vector_type(8)))  float  v8f;

union FragBF16 { v16bf v; u32 u[8]; u16 s[16]; };

static __device__ __forceinline__ u16 f32_to_bf16(float f) {
  u32 u = __float_as_uint(f);
  u32 r = 0x7FFFu + ((u >> 16) & 1u);   // round-to-nearest-even
  u += r;
  return (u16)(u >> 16);
}

static __device__ __forceinline__ float silu(float g) {
  return g / (1.0f + __expf(-g));
}

// ---------------------------------------------------------------------------
// Kernel 1: router. One wave32 per token: logits = x @ rw^T, softmax -> probs.
// (top-8 of 8 experts + renorm == softmax itself, since K == E.)
// ---------------------------------------------------------------------------
__global__ __launch_bounds__(256)
void moe_router_kernel(const float* __restrict__ x,
                       const float* __restrict__ rw,
                       float* __restrict__ probs) {
  const int wave = (blockIdx.x * blockDim.x + threadIdx.x) >> 5;
  const int lane = threadIdx.x & 31;
  if (wave >= TT) return;
  const float* xr = x + (size_t)wave * HH;

  float xreg[HH / 32];                      // 64 regs/lane
#pragma unroll
  for (int j = 0; j < HH / 32; ++j) xreg[j] = xr[lane + j * 32];

  float logits[EE];
#pragma unroll
  for (int e = 0; e < EE; ++e) {
    const float* w = rw + (size_t)e * HH;
    float s = 0.f;
#pragma unroll
    for (int j = 0; j < HH / 32; ++j) s += xreg[j] * w[lane + j * 32];
#pragma unroll
    for (int off = 16; off > 0; off >>= 1) s += __shfl_xor(s, off, 32);
    logits[e] = s;                          // all lanes hold full sum
  }
  float m = logits[0];
#pragma unroll
  for (int e = 1; e < EE; ++e) m = fmaxf(m, logits[e]);
  float sum = 0.f, p[EE];
#pragma unroll
  for (int e = 0; e < EE; ++e) { p[e] = __expf(logits[e] - m); sum += p[e]; }
  if (lane < EE) probs[(size_t)wave * EE + lane] = p[lane] / sum;
}

// ---------------------------------------------------------------------------
// Kernel 2: x fp32 -> bf16 (once; reused by all 8 experts' gate/up GEMMs)
// ---------------------------------------------------------------------------
__global__ __launch_bounds__(256)
void moe_cvt_kernel(const float* __restrict__ in, u16* __restrict__ out,
                    int n) {
  for (int i = blockIdx.x * blockDim.x + threadIdx.x; i < n;
       i += gridDim.x * blockDim.x)
    out[i] = f32_to_bf16(in[i]);
}

// ---------------------------------------------------------------------------
// Fragment builders (LDS -> per-lane WMMA register layouts, wave32).
// Tile rows have LDS_STRIDE(=40) halves = 20 words; data in first 16 words.
// A 16x32 bf16, row = wbase + (lane&15):
//   lane<16 : K = {0..7, 16..23}  -> words {0..3, 8..11} of the row
//   lane>=16: K = {8..15, 24..31} -> words {4..7, 12..15}
// B 32x16 bf16 from K-transposed tile [N][K]:
//   lane<16 : N = lane,    K = {0..7, 16..23}  -> words {0..3, 8..11} of row N
//   lane>=16: N = lane-16, K = {8..15, 24..31} -> words {4..7, 12..15} of row N
// ---------------------------------------------------------------------------
static __device__ __forceinline__ FragBF16 load_a_frag(const u16* ldsA,
                                                       int row, int lane) {
  FragBF16 a;
  const u32* ar = (const u32*)(ldsA + row * LDS_STRIDE);
  const int base = (lane < 16) ? 0 : 4;
#pragma unroll
  for (int j = 0; j < 4; ++j) a.u[j]     = ar[base + j];
#pragma unroll
  for (int j = 0; j < 4; ++j) a.u[4 + j] = ar[base + 8 + j];
  return a;
}

static __device__ __forceinline__ FragBF16 load_b_frag(const u16* ldsBt,
                                                       int nbase, int lane) {
  FragBF16 b;
  const u32* br = (const u32*)(ldsBt + (nbase + (lane & 15)) * LDS_STRIDE);
  const int kb = (lane < 16) ? 0 : 4;
#pragma unroll
  for (int j = 0; j < 4; ++j) b.u[j]     = br[kb + j];
#pragma unroll
  for (int j = 0; j < 4; ++j) b.u[4 + j] = br[kb + 8 + j];
  return b;
}

#define WMMA_BF16(A, B, C)                                                     \
  __builtin_amdgcn_wmma_f32_16x16x32_bf16(false, (A), false, (B), (short)0,    \
                                          (C), false, false)

// ---------------------------------------------------------------------------
// Kernel 3: gate/up projection for all experts.
// grid = (II/64, TT/128, EE), block = 256 (8 waves).
// Per WG: 128-token x 64-I tile; wave w owns tokens [w*16, w*16+16) x 64 cols.
// Double-buffered LDS, one barrier per K-step, 8 WMMAs / wave / K-step.
// Staging is fully vectorized: 2 x b128 (A) + 2 x b128 (per B matrix) global
// loads per thread per K-step, indexed as uniform-base + 32-bit offset.
// ---------------------------------------------------------------------------
__global__ __launch_bounds__(256)
void moe_gateup_kernel(const u16* __restrict__ xb,
                       const float* __restrict__ wg,
                       const float* __restrict__ wu,
                       u16* __restrict__ hb) {
  __shared__ __align__(16) u16 lA [2][128 * LDS_STRIDE];  // 2 x 10 KB
  __shared__ __align__(16) u16 lBg[2][64 * LDS_STRIDE];   // 2 x 5 KB
  __shared__ __align__(16) u16 lBu[2][64 * LDS_STRIDE];   // 2 x 5 KB

  const int tid  = threadIdx.x;
  const int lane = tid & 31;
  const int wav  = tid >> 5;                 // 0..7 -> 16-token M block
  const int i0   = blockIdx.x * 64;          // I tile base
  const int t0   = blockIdx.y * 128;         // token tile base
  const int e    = blockIdx.z;

  const uint4*  xb4 = (const uint4*)xb;      // 8 halves per uint4
  const float4* wg4 = (const float4*)wg;
  const float4* wu4 = (const float4*)wu;
  const u32 wbase4 = (u32)e * (HH * II / 4); // float4 units

  // per-thread staging coordinates (constant over K loop)
  const int aRow = tid >> 1;                 // 2 threads cover one 128-row pair
  const int aC4  = tid & 1;                  //   .. actually: see idx math below
  (void)aRow; (void)aC4;

  auto stage = [&](int buf, int k0) {
    // A: 128 rows x 32 halves = 512 uint4; 2 per thread
#pragma unroll
    for (int j = 0; j < 2; ++j) {
      int idx = j * 256 + tid;               // 0..511
      int row = idx >> 2;                    // 0..127
      int c4  = idx & 3;                     // 0..3
      u32 gi = (u32)(t0 + row) * (HH / 8) + (u32)(k0 >> 3) + c4;
      ((uint4*)lA[buf])[row * (LDS_STRIDE / 8) + c4] = xb4[gi];
    }
    // B: 32(K) x 64(N) fp32 = 512 float4; 2 per thread per matrix;
    //    convert + transpose into [n][k] bf16
#pragma unroll
    for (int j = 0; j < 2; ++j) {
      int idx = j * 256 + tid;               // 0..511
      int k  = idx >> 4;                     // 0..31
      int f4 = idx & 15;                     // 0..15 -> n = 4*f4
      u32 gi = wbase4 + (u32)(k0 + k) * (II / 4) + (u32)(i0 >> 2) + f4;
      float4 g = wg4[gi];
      float4 u = wu4[gi];
      int n = f4 << 2;
      lBg[buf][(n + 0) * LDS_STRIDE + k] = f32_to_bf16(g.x);
      lBg[buf][(n + 1) * LDS_STRIDE + k] = f32_to_bf16(g.y);
      lBg[buf][(n + 2) * LDS_STRIDE + k] = f32_to_bf16(g.z);
      lBg[buf][(n + 3) * LDS_STRIDE + k] = f32_to_bf16(g.w);
      lBu[buf][(n + 0) * LDS_STRIDE + k] = f32_to_bf16(u.x);
      lBu[buf][(n + 1) * LDS_STRIDE + k] = f32_to_bf16(u.y);
      lBu[buf][(n + 2) * LDS_STRIDE + k] = f32_to_bf16(u.z);
      lBu[buf][(n + 3) * LDS_STRIDE + k] = f32_to_bf16(u.w);
    }
  };

  v8f accG[4] = {v8f{}, v8f{}, v8f{}, v8f{}};
  v8f accU[4] = {v8f{}, v8f{}, v8f{}, v8f{}};

  stage(0, 0);
  __syncthreads();

  int buf = 0;
  for (int kt = 0; kt < HH / 32; ++kt) {
    const int nxt = buf ^ 1;
    if (kt + 1 < HH / 32) stage(nxt, (kt + 1) * 32);  // overlaps WMMAs below

    FragBF16 a = load_a_frag(lA[buf], wav * 16 + (lane & 15), lane);
#pragma unroll
    for (int nb = 0; nb < 4; ++nb) {
      FragBF16 bg = load_b_frag(lBg[buf], nb * 16, lane);
      FragBF16 bu = load_b_frag(lBu[buf], nb * 16, lane);
      accG[nb] = WMMA_BF16(a.v, bg.v, accG[nb]);
      accU[nb] = WMMA_BF16(a.v, bu.v, accU[nb]);
    }
    __syncthreads();
    buf = nxt;
  }

  // epilogue: h = silu(gate) * up; D layout: lane<16 -> M=j, lane>=16 -> M=8+j
#pragma unroll
  for (int nb = 0; nb < 4; ++nb) {
    const int col = i0 + nb * 16 + (lane & 15);
#pragma unroll
    for (int j = 0; j < 8; ++j) {
      int tok = t0 + wav * 16 + ((lane < 16) ? j : 8 + j);
      float h = silu(accG[nb][j]) * accU[nb][j];
      hb[(u32)(e * TT + tok) * II + col] = f32_to_bf16(h);
    }
  }
}

// ---------------------------------------------------------------------------
// Kernel 4: down projection + router-weighted accumulation over experts.
// grid = (HH/64, TT/128), block = 256. Expert loop flattened into the
// double-buffered K pipeline (8 experts x 24 K-steps = 192 stages).
// out[t][h] = sum_e probs[t][e] * (h_e @ Wd_e)[t][h]
// ---------------------------------------------------------------------------
__global__ __launch_bounds__(256)
void moe_down_kernel(const u16* __restrict__ hb,
                     const float* __restrict__ wd,
                     const float* __restrict__ probs,
                     float* __restrict__ out) {
  __shared__ __align__(16) u16 lA[2][128 * LDS_STRIDE];   // 2 x 10 KB
  __shared__ __align__(16) u16 lB[2][64 * LDS_STRIDE];    // 2 x 5 KB

  const int tid  = threadIdx.x;
  const int lane = tid & 31;
  const int wav  = tid >> 5;
  const int h0   = blockIdx.x * 64;          // H tile base
  const int t0   = blockIdx.y * 128;         // token tile base

  constexpr int KSTEPS = II / 32;            // 24
  constexpr int NIT    = EE * KSTEPS;        // 192

  const uint4*  hb4 = (const uint4*)hb;      // 8 halves per uint4
  const float4* wd4 = (const float4*)wd;

  auto stage = [&](int buf, int it) {
    const int e2 = it / KSTEPS;
    const int k0 = (it % KSTEPS) * 32;
    // A (h tile): 128 rows x 32 halves = 512 uint4; 2 per thread
#pragma unroll
    for (int j = 0; j < 2; ++j) {
      int idx = j * 256 + tid;
      int row = idx >> 2;
      int c4  = idx & 3;
      u32 gi = (u32)(e2 * TT + t0 + row) * (II / 8) + (u32)(k0 >> 3) + c4;
      ((uint4*)lA[buf])[row * (LDS_STRIDE / 8) + c4] = hb4[gi];
    }
    // B (Wd tile): 32(K) x 64(N) fp32 = 512 float4; 2 per thread
#pragma unroll
    for (int j = 0; j < 2; ++j) {
      int idx = j * 256 + tid;
      int k  = idx >> 4;
      int f4 = idx & 15;
      u32 gi = (u32)e2 * (II * HH / 4) + (u32)(k0 + k) * (HH / 4) +
               (u32)(h0 >> 2) + f4;
      float4 d = wd4[gi];
      int n = f4 << 2;
      lB[buf][(n + 0) * LDS_STRIDE + k] = f32_to_bf16(d.x);
      lB[buf][(n + 1) * LDS_STRIDE + k] = f32_to_bf16(d.y);
      lB[buf][(n + 2) * LDS_STRIDE + k] = f32_to_bf16(d.z);
      lB[buf][(n + 3) * LDS_STRIDE + k] = f32_to_bf16(d.w);
    }
  };

  v8f acc[4]  = {v8f{}, v8f{}, v8f{}, v8f{}};
  v8f part[4] = {v8f{}, v8f{}, v8f{}, v8f{}};

  stage(0, 0);
  __syncthreads();

  int buf = 0;
  for (int it = 0; it < NIT; ++it) {
    const int e  = it / KSTEPS;
    const int kt = it % KSTEPS;
    const int nxt = buf ^ 1;
    if (it + 1 < NIT) stage(nxt, it + 1);

    FragBF16 a = load_a_frag(lA[buf], wav * 16 + (lane & 15), lane);
#pragma unroll
    for (int nb = 0; nb < 4; ++nb) {
      FragBF16 b = load_b_frag(lB[buf], nb * 16, lane);
      part[nb] = WMMA_BF16(a.v, b.v, part[nb]);
    }

    if (kt == KSTEPS - 1) {
      // end of expert: scale partial by per-token router prob, fold into acc
#pragma unroll
      for (int j = 0; j < 8; ++j) {
        int tok = t0 + wav * 16 + ((lane < 16) ? j : 8 + j);
        float p = probs[(u32)tok * EE + e];
#pragma unroll
        for (int nb = 0; nb < 4; ++nb) acc[nb][j] += part[nb][j] * p;
      }
#pragma unroll
      for (int nb = 0; nb < 4; ++nb) part[nb] = v8f{};
    }
    __syncthreads();
    buf = nxt;
  }

#pragma unroll
  for (int nb = 0; nb < 4; ++nb) {
    const int col = h0 + nb * 16 + (lane & 15);
#pragma unroll
    for (int j = 0; j < 8; ++j) {
      int tok = t0 + wav * 16 + ((lane < 16) ? j : 8 + j);
      out[(u32)tok * HH + col] = acc[nb][j];
    }
  }
}

// ---------------------------------------------------------------------------
// Launcher. Workspace: [x_bf16 8MB][probs 64KB][h_bf16 24MB] ~= 33.6 MB
// ---------------------------------------------------------------------------
extern "C" void kernel_launch(void* const* d_in, const int* in_sizes, int n_in,
                              void* d_out, int out_size, void* d_ws,
                              size_t ws_size, hipStream_t stream) {
  (void)in_sizes; (void)n_in; (void)out_size; (void)ws_size;

  const float* x  = (const float*)d_in[0];   // [T, H]
  const float* rw = (const float*)d_in[1];   // [E, H]
  const float* wg = (const float*)d_in[2];   // [E, H, I]
  const float* wu = (const float*)d_in[3];   // [E, H, I]
  const float* wd = (const float*)d_in[4];   // [E, I, H]
  float* out = (float*)d_out;                // [T, H] fp32

  const size_t xb_bytes = (size_t)TT * HH * sizeof(u16);      // 8 MB
  const size_t pr_bytes = (size_t)TT * EE * sizeof(float);    // 64 KB
  u16*   xb    = (u16*)d_ws;
  float* probs = (float*)((char*)d_ws + xb_bytes);
  u16*   hb    = (u16*)((char*)d_ws + xb_bytes + pr_bytes);

  // 1) router softmax probs
  moe_router_kernel<<<TT / 8, 256, 0, stream>>>(x, rw, probs);
  // 2) x -> bf16
  moe_cvt_kernel<<<1024, 256, 0, stream>>>(x, xb, TT * HH);
  // 3) gate/up + silu for all experts
  dim3 g3(II / 64, TT / 128, EE);
  moe_gateup_kernel<<<g3, 256, 0, stream>>>(xb, wg, wu, hb);
  // 4) down proj + weighted sum over experts
  dim3 g4(HH / 64, TT / 128);
  moe_down_kernel<<<g4, 256, 0, stream>>>(hb, wd, probs, out);
}
